// Vicomtech_71451075936997
// MI455X (gfx1250) — compile-verified, matrix operand order checked by reading, not run on hardware
//
#include <hip/hip_runtime.h>
#include <hip/hip_bf16.h>
#include <math.h>

// ---------------------------------------------------------------------------
// MI455X (gfx1250) implementation.
// All GEMMs run on v_wmma_f32_16x16x32_f16 (wave32). Weights are repacked into
// the WMMA B-fragment layout; the GEMM stages its 8KB B panel per K-step into
// LDS with double-buffered GLOBAL_LOAD_ASYNC_TO_LDS_B128 (ASYNCcnt-tracked),
// so B traffic is shared across the block and overlaps WMMA issue.
// pairs@distil_w is factored into two 192-row GEMMs (top/bottom weight halves)
// combined pairwise -> 47x FLOP reduction vs the naive 9216x1546x768 GEMM.
// Attention is flash-style: WMMA QK^T, online softmax, P staged via LDS,
// WMMA P@V against a pre-transposed V.
// ---------------------------------------------------------------------------

typedef _Float16 half_t;
typedef __attribute__((ext_vector_type(16))) _Float16 v16h;
typedef __attribute__((ext_vector_type(8)))  _Float16 v8h;
typedef __attribute__((ext_vector_type(8)))  float    v8f;

constexpr int Bc  = 4;
constexpr int Sc  = 48;
constexpr int Hc  = 768;
constexpr int Ec  = 5;
constexpr int Rc  = 13;
constexpr int FFc = 3072;
constexpr int NHc = 2;
constexpr int NLc = 2;
constexpr int HDc = Hc / NHc;      // 384
constexpr int Tc  = Sc * Sc;       // 2304
constexpr int Mtok = Bc * Tc;      // 9216
constexpr int Ment = Bc * Sc;      // 192
constexpr int KpD  = 800;          // 773 padded to 32
constexpr int KpIO = 800;          // 770 padded to 32
constexpr int NpRT = 896;          // 770 padded to 128

__device__ __forceinline__ v8f vzero() {
  v8f z = {0.f,0.f,0.f,0.f,0.f,0.f,0.f,0.f};
  return z;
}

// Load a 16-half A/B fragment: 8 contiguous halfs at p, 8 at p+16.
// Caller points p at element [row][kstart + 8*(lane>>4)].
__device__ __forceinline__ v16h frag16(const half_t* p) {
  v8h lo = *(const v8h*)(p);
  v8h hi = *(const v8h*)(p + 16);
  v16h r;
#pragma unroll
  for (int t = 0; t < 8; ++t) { r[t] = lo[t]; r[t + 8] = hi[t]; }
  return r;
}

__device__ __forceinline__ v8f wmma_f16(v16h a, v16h b, v8f c) {
  return __builtin_amdgcn_wmma_f32_16x16x32_f16(false, a, false, b, (short)0, c,
                                                false, false);
}

// Async copy 16B global -> LDS (gfx1250 async engine, tracked by ASYNCcnt).
// ldsOff: wave-relative LDS byte offset = low 32 bits of the generic pointer
// (flat->LDS maps by truncating to 32 bits; addrspacecast(3->0) keeps the
// LDS offset in the low word).
__device__ __forceinline__ void async_g2l_b128(unsigned ldsOff,
                                               const void* gsrc) {
  asm volatile("global_load_async_to_lds_b128 %0, %1, off"
               :
               : "v"(ldsOff), "v"((unsigned long long)(size_t)gsrc)
               : "memory");
}

__device__ __forceinline__ void wait_asynccnt0() {
  asm volatile("s_wait_asynccnt 0x0" ::: "memory");
}

// --------------------------- weight repack --------------------------------
// Packed layout: index = ((nt*ktiles + kt)*32 + lane)*16 + e
// element e of lane holds W[kt*32 + kmap(e, lane>>4)][nt*16 + lane&15]
__global__ void pack_w_kernel(const float* __restrict__ W, int ldw, int K, int N,
                              int Kp, int Np, half_t* __restrict__ out) {
  const int ktiles = Kp >> 5;
  const size_t total = (size_t)Kp * (size_t)Np;
  for (size_t idx = (size_t)blockIdx.x * blockDim.x + threadIdx.x; idx < total;
       idx += (size_t)gridDim.x * blockDim.x) {
    int e    = (int)(idx & 15);
    int lane = (int)((idx >> 4) & 31);
    size_t t = idx >> 9;
    int kt = (int)(t % ktiles);
    int nt = (int)(t / ktiles);
    int hi = lane >> 4;
    int k  = kt * 32 + ((e < 8) ? e : e + 8) + 8 * hi;
    int n  = nt * 16 + (lane & 15);
    float v = (k < K && n < N) ? W[(size_t)k * ldw + n] : 0.f;
    out[idx] = (half_t)v;
  }
}

// ----------------------------- WMMA GEMM ----------------------------------
// C[M,N] = act(A[M,Kp](f16) @ Bpacked + bias) * alpha
// ACT: 0 none, 1 tanh, 2 gelu(exact erf), 3 mish
// B panel (8 n-tiles x 512 halfs = 8KB per K-step) is double-buffered in LDS
// via async global->LDS copies; A fragments are direct 2xb128 global loads.
template <int ACT, bool OUTH>
__global__ __launch_bounds__(256) void gemm_wmma(
    const half_t* __restrict__ A, int Kp, const half_t* __restrict__ Bp,
    const float* __restrict__ bias, int M, int N, int ldc, float alpha,
    float* __restrict__ Cf, half_t* __restrict__ Ch) {
  const int lane = threadIdx.x & 31;
  const int wave = threadIdx.x >> 5;
  const int wm = wave >> 2, wn = wave & 3;
  const int rowBase = blockIdx.y * 64 + wm * 32;
  const int colBase = blockIdx.x * 128 + wn * 32;
  const int ntb = (int)(blockIdx.x * 128) >> 4;  // block's base n-tile
  const int ktiles = Kp >> 5;
  const int lrow = lane & 15;
  const int hi8 = (lane >> 4) << 3;

  __shared__ half_t bsm[2][8 * 512];  // 2 x 8KB B panels

  v8f acc[2][2];
  acc[0][0] = vzero(); acc[0][1] = vzero();
  acc[1][0] = vzero(); acc[1][1] = vzero();

  const half_t* a0 = A + (size_t)(rowBase + lrow) * Kp + hi8;
  const half_t* a1 = a0 + (size_t)16 * Kp;

  // Stage B panel for step kt into buffer buf: each of 256 threads issues two
  // async b128 copies (256*2*16B = 8KB).
  auto stage = [&](int kt, int buf) {
#pragma unroll
    for (int c = 0; c < 2; ++c) {
      unsigned hIdx = (threadIdx.x * 2u + c) * 8u;  // half index in panel
      int ntL = (int)(hIdx >> 9);
      unsigned rem = hIdx & 511u;
      const half_t* src = Bp + ((size_t)(ntb + ntL) * ktiles + kt) * 512 + rem;
      unsigned dst = (unsigned)(size_t)(const void*)&bsm[buf][hIdx];
      async_g2l_b128(dst, src);
    }
  };

  stage(0, 0);
  for (int kt = 0; kt < ktiles; ++kt) {
    wait_asynccnt0();   // own-wave async copies for buf[kt&1] done
    __syncthreads();    // ...and everyone else's too
    if (kt + 1 < ktiles) stage(kt + 1, (kt + 1) & 1);
    // (no trailing barrier needed: buf[(kt+1)&1] was last read in iteration
    //  kt-1, and every wave passed the barrier above after finishing it)

    v16h af0 = frag16(a0 + kt * 32);
    v16h af1 = frag16(a1 + kt * 32);
    const half_t* bw = &bsm[kt & 1][0] + (size_t)(wn * 2) * 512 + lane * 16;
    v16h bf0 = *(const v16h*)(bw);
    v16h bf1 = *(const v16h*)(bw + 512);
    acc[0][0] = wmma_f16(af0, bf0, acc[0][0]);
    acc[0][1] = wmma_f16(af0, bf1, acc[0][1]);
    acc[1][0] = wmma_f16(af1, bf0, acc[1][0]);
    acc[1][1] = wmma_f16(af1, bf1, acc[1][1]);
  }

#pragma unroll
  for (int mi = 0; mi < 2; ++mi) {
#pragma unroll
    for (int ni = 0; ni < 2; ++ni) {
      int col = colBase + ni * 16 + lrow;
      float bv = (bias != nullptr && col < N) ? bias[col] : 0.f;
#pragma unroll
      for (int r = 0; r < 8; ++r) {
        int row = rowBase + mi * 16 + r + hi8;
        if (row < M && col < N) {
          float v = acc[mi][ni][r] + bv;
          if (ACT == 1) v = tanhf(v);
          else if (ACT == 2) v = 0.5f * v * (1.0f + erff(v * 0.70710678118654752f));
          else if (ACT == 3) {
            float sp = (v > 20.f) ? v : log1pf(__expf(v));
            v = v * tanhf(sp);
          }
          v *= alpha;
          size_t o = (size_t)row * ldc + col;
          if (OUTH) Ch[o] = (half_t)v;
          else      Cf[o] = v;
        }
      }
    }
  }
}

// -------------------------- flash attention -------------------------------
// grid: (Tc/64, Bc*NHc), block 256. Wave = (qgroup 0..3) x (hd-half 0..1).
__global__ __launch_bounds__(256) void attn_kernel(
    const half_t* __restrict__ q16, const half_t* __restrict__ k16,
    const half_t* __restrict__ vt16, half_t* __restrict__ ctx16) {
  const int lane = threadIdx.x & 31;
  const int wave = threadIdx.x >> 5;
  const int qg = wave >> 1;
  const int hh = wave & 1;
  const int bh = blockIdx.y;
  const int bb = bh >> 1;
  const int hidx = bh & 1;
  const int hd0 = hidx * HDc;
  const int qbase = blockIdx.x * 64 + qg * 16;
  const int lrow = lane & 15;
  const int hi8 = (lane >> 4) << 3;

  __shared__ half_t plds[8 * 16 * 64];
  half_t* pw = plds + wave * (16 * 64);  // wave-private: no block barrier needed

  const half_t* qrow = q16 + ((size_t)(bb * Tc + qbase + lrow) * Hc + hd0) + hi8;
  const half_t* kbaseP = k16 + (size_t)bb * Tc * Hc + hd0 + hi8;
  const half_t* vbaseP = vt16 + ((size_t)bb * NHc + hidx) * (size_t)HDc * Tc;

  float mrun[8], lrun[8];
#pragma unroll
  for (int r = 0; r < 8; ++r) { mrun[r] = -3.0e38f; lrun[r] = 0.f; }
  v8f oacc[12];
#pragma unroll
  for (int n = 0; n < 12; ++n) oacc[n] = vzero();

  for (int jt = 0; jt < Tc / 64; ++jt) {
    const int keybase = jt * 64;
    v8f sacc[4];
#pragma unroll
    for (int n = 0; n < 4; ++n) sacc[n] = vzero();

    for (int kk = 0; kk < HDc / 32; ++kk) {
      v16h af = frag16(qrow + kk * 32);
#pragma unroll
      for (int ns = 0; ns < 4; ++ns) {
        const half_t* kp = kbaseP + (size_t)(keybase + ns * 16 + lrow) * Hc + kk * 32;
        v16h bf = frag16(kp);
        sacc[ns] = wmma_f16(af, bf, sacc[ns]);
      }
    }

    float rmax[8];
#pragma unroll
    for (int r = 0; r < 8; ++r) {
      float m = fmaxf(fmaxf(sacc[0][r], sacc[1][r]), fmaxf(sacc[2][r], sacc[3][r]));
      rmax[r] = m;
    }
#pragma unroll
    for (int d = 1; d < 16; d <<= 1) {
#pragma unroll
      for (int r = 0; r < 8; ++r)
        rmax[r] = fmaxf(rmax[r], __shfl_xor(rmax[r], d, 32));
    }
    float al[8], psum[8];
#pragma unroll
    for (int r = 0; r < 8; ++r) {
      float nm = fmaxf(mrun[r], rmax[r]);
      al[r] = __expf(mrun[r] - nm);
      mrun[r] = nm;
      psum[r] = 0.f;
    }
#pragma unroll
    for (int ns = 0; ns < 4; ++ns) {
#pragma unroll
      for (int r = 0; r < 8; ++r) {
        float p = __expf(sacc[ns][r] - mrun[r]);
        psum[r] += p;
        pw[(r + hi8) * 64 + ns * 16 + lrow] = (half_t)p;
      }
    }
#pragma unroll
    for (int d = 1; d < 16; d <<= 1) {
#pragma unroll
      for (int r = 0; r < 8; ++r) psum[r] += __shfl_xor(psum[r], d, 32);
    }
#pragma unroll
    for (int r = 0; r < 8; ++r) lrun[r] = lrun[r] * al[r] + psum[r];
#pragma unroll
    for (int n = 0; n < 12; ++n) {
#pragma unroll
      for (int r = 0; r < 8; ++r) oacc[n][r] *= al[r];
    }
    // P @ V (LDS is in-order within a wave: stores above are visible)
#pragma unroll
    for (int k2 = 0; k2 < 2; ++k2) {
      v16h pf = frag16(pw + lrow * 64 + k2 * 32 + hi8);
#pragma unroll
      for (int nt = 0; nt < 12; ++nt) {
        const half_t* vp = vbaseP + (size_t)(hh * 192 + nt * 16 + lrow) * Tc +
                           keybase + k2 * 32 + hi8;
        v16h vf = frag16(vp);
        oacc[nt] = wmma_f16(pf, vf, oacc[nt]);
      }
    }
  }
#pragma unroll
  for (int nt = 0; nt < 12; ++nt) {
#pragma unroll
    for (int r = 0; r < 8; ++r) {
      int row = qbase + r + hi8;
      int hd = hh * 192 + nt * 16 + lrow;
      ctx16[(size_t)(bb * Tc + row) * Hc + hd0 + hd] = (half_t)(oacc[nt][r] / lrun[r]);
    }
  }
}

// ----------------------------- small kernels ------------------------------

__global__ void cvt_pad_kernel(const float* __restrict__ src, int M, int K, int Kp,
                               half_t* __restrict__ dst) {
  const size_t total = (size_t)M * Kp;
  for (size_t i = (size_t)blockIdx.x * blockDim.x + threadIdx.x; i < total;
       i += (size_t)gridDim.x * blockDim.x) {
    size_t row = i / Kp;
    int c = (int)(i % Kp);
    dst[i] = (c < K) ? (half_t)src[row * K + c] : (half_t)0.f;
  }
}

__global__ void build_ee_kernel(const float* __restrict__ emb,
                                const float* __restrict__ logE,
                                half_t* __restrict__ ee) {
  const size_t total = (size_t)Ment * KpD;
  for (size_t i = (size_t)blockIdx.x * blockDim.x + threadIdx.x; i < total;
       i += (size_t)gridDim.x * blockDim.x) {
    size_t row = i / KpD;
    int c = (int)(i % KpD);
    float v = 0.f;
    if (c < Hc) v = emb[row * Hc + c];
    else if (c < Hc + Ec) v = logE[row * Ec + (c - Hc)];
    ee[i] = (half_t)v;
  }
}

__global__ void build_io_kernel(const half_t* __restrict__ x16,
                                const float* __restrict__ logR,
                                half_t* __restrict__ io) {
  const size_t total = (size_t)Mtok * KpIO;
  for (size_t i = (size_t)blockIdx.x * blockDim.x + threadIdx.x; i < total;
       i += (size_t)gridDim.x * blockDim.x) {
    size_t row = i / KpIO;
    int c = (int)(i % KpIO);
    half_t v = (half_t)0.f;
    if (c < Hc) v = x16[row * Hc + c];
    else if (c < Hc + 2) v = (half_t)logR[row * 2 + (c - Hc)];
    io[i] = v;
  }
}

__global__ void vtrans_kernel(const half_t* __restrict__ v16,
                              half_t* __restrict__ vt16) {
  const size_t total = (size_t)Mtok * Hc;
  for (size_t i = (size_t)blockIdx.x * blockDim.x + threadIdx.x; i < total;
       i += (size_t)gridDim.x * blockDim.x) {
    size_t bt = i / Hc;
    int ch = (int)(i % Hc);
    int b = (int)(bt / Tc);
    int t = (int)(bt % Tc);
    int h = ch / HDc;
    int d = ch % HDc;
    vt16[((size_t)(b * NHc + h) * HDc + d) * Tc + t] = v16[i];
  }
}

// x = LN(tanh(A1[i] + A2[j] + bd) + pos)  (distil + emb_ln, fused pairwise)
__global__ void pairs_ln_kernel(const float* __restrict__ A1,
                                const float* __restrict__ A2,
                                const float* __restrict__ bd,
                                const float* __restrict__ pos,
                                const float* __restrict__ g,
                                const float* __restrict__ be,
                                float* __restrict__ x, half_t* __restrict__ x16) {
  const int b = blockIdx.y;
  const int t = blockIdx.x;
  const int i = t / Sc, j = t % Sc;
  const float* a1 = A1 + (size_t)(b * Sc + i) * Hc;
  const float* a2 = A2 + (size_t)(b * Sc + j) * Hc;
  const int tid = threadIdx.x;
  float v[3];
  float s = 0.f, sq = 0.f;
#pragma unroll
  for (int u = 0; u < 3; ++u) {
    int c = tid + u * 256;
    float w = tanhf(a1[c] + a2[c] + bd[c]) + pos[(size_t)t * Hc + c];
    v[u] = w; s += w; sq += w * w;
  }
  __shared__ float r1[256], r2[256];
  r1[tid] = s; r2[tid] = sq;
  __syncthreads();
  for (int st = 128; st; st >>= 1) {
    if (tid < st) { r1[tid] += r1[tid + st]; r2[tid] += r2[tid + st]; }
    __syncthreads();
  }
  float mean = r1[0] * (1.f / Hc);
  float var = r2[0] * (1.f / Hc) - mean * mean;
  float inv = rsqrtf(var + 1e-12f);
  size_t rb = (size_t)(b * Tc + t) * Hc;
#pragma unroll
  for (int u = 0; u < 3; ++u) {
    int c = tid + u * 256;
    float o = (v[u] - mean) * inv * g[c] + be[c];
    x[rb + c] = o;
    x16[rb + c] = (half_t)o;
  }
}

// x = LN(x + y); also writes x16
__global__ void add_ln_kernel(float* __restrict__ x, const float* __restrict__ y,
                              const float* __restrict__ g,
                              const float* __restrict__ be,
                              half_t* __restrict__ x16) {
  const size_t row = blockIdx.x;
  float* xr = x + row * Hc;
  const float* yr = y + row * Hc;
  const int tid = threadIdx.x;
  float v[3];
  float s = 0.f, sq = 0.f;
#pragma unroll
  for (int u = 0; u < 3; ++u) {
    int c = tid + u * 256;
    float w = xr[c] + yr[c];
    v[u] = w; s += w; sq += w * w;
  }
  __shared__ float r1[256], r2[256];
  r1[tid] = s; r2[tid] = sq;
  __syncthreads();
  for (int st = 128; st; st >>= 1) {
    if (tid < st) { r1[tid] += r1[tid + st]; r2[tid] += r2[tid + st]; }
    __syncthreads();
  }
  float mean = r1[0] * (1.f / Hc);
  float var = r2[0] * (1.f / Hc) - mean * mean;
  float inv = rsqrtf(var + 1e-12f);
#pragma unroll
  for (int u = 0; u < 3; ++u) {
    int c = tid + u * 256;
    float o = (v[u] - mean) * inv * g[c] + be[c];
    xr[c] = o;
    x16[row * Hc + c] = (half_t)o;
  }
}

// logits = h @ W2 + b2; out = log_softmax(logits); optional raw logits
template <int C>
__global__ void cls_head_kernel(const float* __restrict__ h, int ld, int Hd,
                                const float* __restrict__ W2,
                                const float* __restrict__ b2,
                                float* __restrict__ outLS,
                                float* __restrict__ raw) {
  const size_t row = blockIdx.x;
  const float* hr = h + row * (size_t)ld;
  const int tid = threadIdx.x;
  float part[C];
#pragma unroll
  for (int c = 0; c < C; ++c) part[c] = 0.f;
  for (int c0 = tid; c0 < Hd; c0 += blockDim.x) {
    float hv = hr[c0];
#pragma unroll
    for (int c = 0; c < C; ++c) part[c] += hv * W2[(size_t)c0 * C + c];
  }
  __shared__ float red[256];
  float logit[C];
#pragma unroll
  for (int c = 0; c < C; ++c) {
    red[tid] = part[c];
    __syncthreads();
    for (int st = 128; st; st >>= 1) {
      if (tid < st) red[tid] += red[tid + st];
      __syncthreads();
    }
    logit[c] = red[0] + b2[c];
    __syncthreads();
  }
  float mx = logit[0];
#pragma unroll
  for (int c = 1; c < C; ++c) mx = fmaxf(mx, logit[c]);
  float sum = 0.f;
#pragma unroll
  for (int c = 0; c < C; ++c) sum += __expf(logit[c] - mx);
  float lse = mx + __logf(sum);
  if (tid < C) {
    if (raw) raw[row * C + tid] = logit[tid];
    outLS[row * C + tid] = logit[tid] - lse;
  }
}

// ------------------------------- host -------------------------------------

static inline size_t align_up(size_t v) { return (v + 255) & ~(size_t)255; }

extern "C" void kernel_launch(void* const* d_in, const int* in_sizes, int n_in,
                              void* d_out, int out_size, void* d_ws, size_t ws_size,
                              hipStream_t stream) {
  (void)in_sizes; (void)n_in; (void)out_size; (void)ws_size;
  auto F = [&](int i) { return (const float*)d_in[i]; };

  const float* emb   = F(0);
  const float* ent1w = F(1);  const float* ent1b = F(2);
  const float* ent2w = F(3);  const float* ent2b = F(4);
  const float* distw = F(5);  const float* distb = F(6);
  const float* pos   = F(7);
  const float* elng  = F(8);  const float* elnb  = F(9);
  struct LayerP {
    const float *qw, *qb, *kw, *kb, *vw, *vb, *ow, *ob;
    const float *g1, *b1, *f1w, *f1b, *f2w, *f2b, *g2, *b2;
  } L[NLc];
  int idx = 10;
  for (int l = 0; l < NLc; ++l) {
    L[l].qw = F(idx + 0);  L[l].qb = F(idx + 1);
    L[l].kw = F(idx + 2);  L[l].kb = F(idx + 3);
    L[l].vw = F(idx + 4);  L[l].vb = F(idx + 5);
    L[l].ow = F(idx + 6);  L[l].ob = F(idx + 7);
    L[l].g1 = F(idx + 8);  L[l].b1 = F(idx + 9);
    L[l].f1w = F(idx + 10); L[l].f1b = F(idx + 11);
    L[l].f2w = F(idx + 12); L[l].f2b = F(idx + 13);
    L[l].g2 = F(idx + 14); L[l].b2 = F(idx + 15);
    idx += 16;
  }
  const float* mw1w = F(42); const float* mw1b = F(43);
  const float* mw2w = F(44); const float* mw2b = F(45);
  const float* sa1w = F(46); const float* sa1b = F(47);
  const float* sa2w = F(48); const float* sa2b = F(49);
  const float* re1w = F(50); const float* re1b = F(51);
  const float* re2w = F(52); const float* re2b = F(53);
  const float* rt1w = F(54); const float* rt1b = F(55);
  const float* rt2w = F(56); const float* rt2b = F(57);

  float* out = (float*)d_out;
  float* outEnt = out + 0;
  float* outMW  = out + 960;
  float* outSA  = out + 960 + 18432;
  float* outREL = out + 960 + 2 * 18432;
  float* outRT  = out + 960 + 3 * 18432;

  // ---- workspace bump allocator ----
  char* base = (char*)d_ws;
  size_t off = 0;
  auto allocB = [&](size_t bytes) -> void* {
    off = align_up(off);
    void* p = base + off;
    off += bytes;
    return p;
  };
  auto allocH = [&](size_t n) { return (half_t*)allocB(n * sizeof(half_t)); };
  auto allocF = [&](size_t n) { return (float*)allocB(n * sizeof(float)); };

  // packed weights
  half_t* pEnt1 = allocH((size_t)Hc * Hc);
  half_t* pDT = allocH((size_t)KpD * Hc);
  half_t* pDB = allocH((size_t)KpD * Hc);
  half_t *pQ[NLc], *pK[NLc], *pV[NLc], *pO[NLc], *pF1[NLc], *pF2[NLc];
  for (int l = 0; l < NLc; ++l) {
    pQ[l] = allocH((size_t)Hc * Hc);
    pK[l] = allocH((size_t)Hc * Hc);
    pV[l] = allocH((size_t)Hc * Hc);
    pO[l] = allocH((size_t)Hc * Hc);
    pF1[l] = allocH((size_t)Hc * FFc);
    pF2[l] = allocH((size_t)FFc * Hc);
  }
  half_t* pMW = allocH((size_t)Hc * Hc);
  half_t* pSA = allocH((size_t)Hc * Hc);
  half_t* pRE = allocH((size_t)Hc * Hc);
  half_t* pRT = allocH((size_t)KpIO * NpRT);

  // activations
  half_t* emb16 = allocH((size_t)Ment * Hc);
  float*  hEnt  = allocF((size_t)Ment * Hc);
  float*  logE  = allocF((size_t)Ment * 8);
  half_t* ee16  = allocH((size_t)Ment * KpD);
  float*  A1    = allocF((size_t)Ment * Hc);
  float*  A2    = allocF((size_t)Ment * Hc);
  float*  xbuf  = allocF((size_t)Mtok * Hc);
  half_t* x16   = allocH((size_t)Mtok * Hc);
  half_t* q16   = allocH((size_t)Mtok * Hc);
  half_t* k16   = allocH((size_t)Mtok * Hc);
  half_t* vtmp  = allocH((size_t)Mtok * Hc);
  half_t* vt16  = allocH((size_t)Mtok * Hc);
  half_t* ctx16 = allocH((size_t)Mtok * Hc);
  float*  ytmp  = allocF((size_t)Mtok * 784);  // reused: O/FF2 out, cls hidden, rt hidden
  half_t* ff16  = allocH((size_t)Mtok * FFc);  // reused as io16
  float*  logR  = allocF((size_t)Mtok * 2);
  half_t* io16  = ff16;

  const dim3 blk(256);
  const dim3 gsS(1024);
  const float qscale = 1.0f / sqrtf((float)HDc);

  // ---- pack all weights (zero-padded into WMMA fragment layout) ----
  pack_w_kernel<<<gsS, blk, 0, stream>>>(ent1w, Hc, Hc, Hc, Hc, Hc, pEnt1);
  pack_w_kernel<<<gsS, blk, 0, stream>>>(distw, Hc, 773, Hc, KpD, Hc, pDT);
  pack_w_kernel<<<gsS, blk, 0, stream>>>(distw + (size_t)773 * Hc, Hc, 773, Hc, KpD, Hc, pDB);
  for (int l = 0; l < NLc; ++l) {
    pack_w_kernel<<<gsS, blk, 0, stream>>>(L[l].qw, Hc, Hc, Hc, Hc, Hc, pQ[l]);
    pack_w_kernel<<<gsS, blk, 0, stream>>>(L[l].kw, Hc, Hc, Hc, Hc, Hc, pK[l]);
    pack_w_kernel<<<gsS, blk, 0, stream>>>(L[l].vw, Hc, Hc, Hc, Hc, Hc, pV[l]);
    pack_w_kernel<<<gsS, blk, 0, stream>>>(L[l].ow, Hc, Hc, Hc, Hc, Hc, pO[l]);
    pack_w_kernel<<<gsS, blk, 0, stream>>>(L[l].f1w, FFc, Hc, FFc, Hc, FFc, pF1[l]);
    pack_w_kernel<<<gsS, blk, 0, stream>>>(L[l].f2w, Hc, FFc, Hc, FFc, Hc, pF2[l]);
  }
  pack_w_kernel<<<gsS, blk, 0, stream>>>(mw1w, Hc, Hc, Hc, Hc, Hc, pMW);
  pack_w_kernel<<<gsS, blk, 0, stream>>>(sa1w, Hc, Hc, Hc, Hc, Hc, pSA);
  pack_w_kernel<<<gsS, blk, 0, stream>>>(re1w, Hc, Hc, Hc, Hc, Hc, pRE);
  pack_w_kernel<<<gsS, blk, 0, stream>>>(rt1w, 770, 770, 770, KpIO, NpRT, pRT);

  // ---- entity classifier ----
  cvt_pad_kernel<<<gsS, blk, 0, stream>>>(emb, Ment, Hc, Hc, emb16);
  gemm_wmma<3, false><<<dim3(Hc / 128, Ment / 64), blk, 0, stream>>>(
      emb16, Hc, pEnt1, ent1b, Ment, Hc, Hc, 1.f, hEnt, (half_t*)nullptr);
  cls_head_kernel<Ec><<<Ment, blk, 0, stream>>>(hEnt, Hc, Hc, ent2w, ent2b, outEnt, logE);

  // ---- pairs + distil (factored) + emb LayerNorm ----
  build_ee_kernel<<<gsS, blk, 0, stream>>>(emb, logE, ee16);
  gemm_wmma<0, false><<<dim3(Hc / 128, Ment / 64), blk, 0, stream>>>(
      ee16, KpD, pDT, (const float*)nullptr, Ment, Hc, Hc, 1.f, A1, (half_t*)nullptr);
  gemm_wmma<0, false><<<dim3(Hc / 128, Ment / 64), blk, 0, stream>>>(
      ee16, KpD, pDB, (const float*)nullptr, Ment, Hc, Hc, 1.f, A2, (half_t*)nullptr);
  pairs_ln_kernel<<<dim3(Tc, Bc), blk, 0, stream>>>(A1, A2, distb, pos, elng, elnb,
                                                    xbuf, x16);

  // ---- transformer layers ----
  const dim3 gGH(Hc / 128, Mtok / 64);
  const dim3 gGF(FFc / 128, Mtok / 64);
  for (int l = 0; l < NLc; ++l) {
    gemm_wmma<0, true><<<gGH, blk, 0, stream>>>(x16, Hc, pQ[l], L[l].qb, Mtok, Hc,
                                                Hc, qscale, (float*)nullptr, q16);
    gemm_wmma<0, true><<<gGH, blk, 0, stream>>>(x16, Hc, pK[l], L[l].kb, Mtok, Hc,
                                                Hc, 1.f, (float*)nullptr, k16);
    gemm_wmma<0, true><<<gGH, blk, 0, stream>>>(x16, Hc, pV[l], L[l].vb, Mtok, Hc,
                                                Hc, 1.f, (float*)nullptr, vtmp);
    vtrans_kernel<<<gsS, blk, 0, stream>>>(vtmp, vt16);
    attn_kernel<<<dim3(Tc / 64, Bc * NHc), blk, 0, stream>>>(q16, k16, vt16, ctx16);
    gemm_wmma<0, false><<<gGH, blk, 0, stream>>>(ctx16, Hc, pO[l], L[l].ob, Mtok, Hc,
                                                 Hc, 1.f, ytmp, (half_t*)nullptr);
    add_ln_kernel<<<Mtok, blk, 0, stream>>>(xbuf, ytmp, L[l].g1, L[l].b1, x16);
    gemm_wmma<2, true><<<gGF, blk, 0, stream>>>(x16, Hc, pF1[l], L[l].f1b, Mtok, FFc,
                                                FFc, 1.f, (float*)nullptr, ff16);
    gemm_wmma<0, false><<<gGH, blk, 0, stream>>>(ff16, FFc, pF2[l], L[l].f2b, Mtok,
                                                 Hc, Hc, 1.f, ytmp, (half_t*)nullptr);
    add_ln_kernel<<<Mtok, blk, 0, stream>>>(xbuf, ytmp, L[l].g2, L[l].b2, x16);
  }

  // ---- output classifiers ----
  gemm_wmma<3, false><<<gGH, blk, 0, stream>>>(x16, Hc, pMW, mw1b, Mtok, Hc, Hc,
                                               1.f, ytmp, (half_t*)nullptr);
  cls_head_kernel<2><<<Mtok, blk, 0, stream>>>(ytmp, Hc, Hc, mw2w, mw2b, outMW,
                                               (float*)nullptr);
  gemm_wmma<3, false><<<gGH, blk, 0, stream>>>(x16, Hc, pSA, sa1b, Mtok, Hc, Hc,
                                               1.f, ytmp, (half_t*)nullptr);
  cls_head_kernel<2><<<Mtok, blk, 0, stream>>>(ytmp, Hc, Hc, sa2w, sa2b, outSA,
                                               (float*)nullptr);
  gemm_wmma<3, false><<<gGH, blk, 0, stream>>>(x16, Hc, pRE, re1b, Mtok, Hc, Hc,
                                               1.f, ytmp, (half_t*)nullptr);
  cls_head_kernel<2><<<Mtok, blk, 0, stream>>>(ytmp, Hc, Hc, re2w, re2b, outREL, logR);

  build_io_kernel<<<gsS, blk, 0, stream>>>(x16, logR, io16);
  gemm_wmma<3, false><<<dim3(NpRT / 128, Mtok / 64), blk, 0, stream>>>(
      io16, KpIO, pRT, rt1b, Mtok, 770, 770, 1.f, ytmp, (half_t*)nullptr);
  cls_head_kernel<Rc><<<Mtok, blk, 0, stream>>>(ytmp, 770, 770, rt2w, rt2b, outRT,
                                                (float*)nullptr);
}
